// MoDSegmentRouter_61615600828933
// MI455X (gfx1250) — matrix operand order repaired and checked by reading, not compiled, not run on hardware
//
#include <hip/hip_runtime.h>
#include <hip/hip_bf16.h>
#include <math.h>

typedef __attribute__((ext_vector_type(16))) _Float16 v16h;
typedef __attribute__((ext_vector_type(8)))  float    v8f;

#define D_MODEL 2048
#define HIDDEN  512
#define BATCH   4
#define SEQ     4096
#define TOPK    2048   // SEQ * 0.5

// ---------------------------------------------------------------------------
// Kernel 1: convert W1 [D,H] f32 -> W1T [H,D] f16 (transposed so the WMMA
// B-fragment per lane is 16 contiguous halfs = one 32B load).
// ---------------------------------------------------------------------------
__global__ __launch_bounds__(256) void k_convert_w1(const float* __restrict__ W1,
                                                    _Float16* __restrict__ W1T) {
  int idx = blockIdx.x * 256 + threadIdx.x;     // over H*D (output layout)
  if (idx >= D_MODEL * HIDDEN) return;
  int h = idx / D_MODEL;
  int d = idx - h * D_MODEL;
  W1T[idx] = (_Float16)W1[d * HIDDEN + h];
}

// ---------------------------------------------------------------------------
// Kernel 2: init scores to b2 (the fused GEMM kernel atomically accumulates).
// ---------------------------------------------------------------------------
__global__ __launch_bounds__(256) void k_init_scores(float* __restrict__ scores,
                                                     const float* __restrict__ b2) {
  int i = blockIdx.x * 256 + threadIdx.x;
  if (i < BATCH * SEQ) scores[i] = b2[0];
}

// ---------------------------------------------------------------------------
// Kernel 3: fused router scores.
//   scores[row] += sum_h gelu( (x @ W1)[row,h] + b1[h] ) * W2[h]
// Grid: BATCH*SEQ/16 blocks of 256 threads (8 waves).
// Wave w owns token-tile [blockIdx.x*16,+16) x hidden cols [w*64,+64) as 4
// column tiles of 16. K-loop over D in steps of 32 with v_wmma_f32_16x16x32_f16,
// A-fragment reuse across the 4 tiles, and explicit double-buffered loads so
// next-step global traffic overlaps current-step WMMAs.
// ---------------------------------------------------------------------------
__global__ __launch_bounds__(256) void k_router_scores(
    const float*    __restrict__ x,      // [B*S, D] f32
    const _Float16* __restrict__ W1T,    // [H, D]   f16
    const float*    __restrict__ b1,     // [H]
    const float*    __restrict__ W2,     // [H]
    float*          __restrict__ scores) // [B*S]
{
  const int tid      = threadIdx.x;
  const int lane     = tid & 31;
  const int wave     = tid >> 5;
  const int rowBase  = blockIdx.x * 16;
  const int colTile0 = wave * 4;

  // A-fragment addressing (16-bit A 16x32, contiguous k order used here, same as B):
  //   lanes 0-15: row M=lane,    K in {kb..kb+15}, kb=0
  //   lanes16-31: row M=lane-16, same pattern with kb=16
  const int m      = lane & 15;
  const int half   = lane >> 4;
  const int kbaseA = half * 16;
  // B-fragment addressing (16-bit B 32x16): lane&15 = N, lane half picks
  // K range 0-15 / 16-31; 16 consecutive halfs per lane.
  const int n     = lane & 15;
  const int koffB = half * 16;

  const float* px = x + (size_t)(rowBase + m) * D_MODEL;

  v8f c[4];
  for (int t = 0; t < 4; ++t)
    for (int e = 0; e < 8; ++e) c[t][e] = 0.0f;

  const _Float16* pb[4];
  for (int t = 0; t < 4; ++t) {
    int hcol = (colTile0 + t) * 16 + n;
    pb[t] = W1T + (size_t)hcol * D_MODEL + koffB;
  }

  // ---- double-buffered fragments: [2] ping-pong register sets
  float4 fA[2][4];
  v16h   fB[2][4];

  auto loadStep = [&](int kt, int buf) {
    fA[buf][0] = *(const float4*)(px + kt + kbaseA + 0);
    fA[buf][1] = *(const float4*)(px + kt + kbaseA + 4);
    fA[buf][2] = *(const float4*)(px + kt + kbaseA + 8);
    fA[buf][3] = *(const float4*)(px + kt + kbaseA + 12);
    fB[buf][0] = *(const v16h*)(pb[0] + kt);
    fB[buf][1] = *(const v16h*)(pb[1] + kt);
    fB[buf][2] = *(const v16h*)(pb[2] + kt);
    fB[buf][3] = *(const v16h*)(pb[3] + kt);
  };

  auto compute = [&](int buf) {
    v16h a;
    a[0]  = (_Float16)fA[buf][0].x; a[1]  = (_Float16)fA[buf][0].y;
    a[2]  = (_Float16)fA[buf][0].z; a[3]  = (_Float16)fA[buf][0].w;
    a[4]  = (_Float16)fA[buf][1].x; a[5]  = (_Float16)fA[buf][1].y;
    a[6]  = (_Float16)fA[buf][1].z; a[7]  = (_Float16)fA[buf][1].w;
    a[8]  = (_Float16)fA[buf][2].x; a[9]  = (_Float16)fA[buf][2].y;
    a[10] = (_Float16)fA[buf][2].z; a[11] = (_Float16)fA[buf][2].w;
    a[12] = (_Float16)fA[buf][3].x; a[13] = (_Float16)fA[buf][3].y;
    a[14] = (_Float16)fA[buf][3].z; a[15] = (_Float16)fA[buf][3].w;
    for (int t = 0; t < 4; ++t) {
      c[t] = __builtin_amdgcn_wmma_f32_16x16x32_f16(
          /*neg_a=*/false, a, /*neg_b=*/false, fB[buf][t],
          /*c_mod=*/(short)0, c[t], /*reuse_a=*/false, /*reuse_b=*/false);
    }
  };

  loadStep(0, 0);                       // prologue
  for (int kt = 0; kt < D_MODEL; kt += 64) {
    // body A: compute buffer 0 (K = kt), prefetch buffer 1 (K = kt+32)
    loadStep(kt + 32, 1);
    compute(0);
    // body B: compute buffer 1 (K = kt+32), prefetch buffer 0 (K = kt+64)
    int ktn = (kt + 64 < D_MODEL) ? (kt + 64) : 0;   // harmless dummy on last
    loadStep(ktn, 0);
    compute(1);
  }

  // ---- Epilogue: gelu(.) * W2, reduce across N (16 lanes of a half),
  // one atomicAdd per output row. C layout: VGPR g -> row g + 8*half, N=lane&15.
  for (int g = 0; g < 8; ++g) {
    float acc = 0.0f;
    for (int t = 0; t < 4; ++t) {
      int hcol  = (colTile0 + t) * 16 + n;
      float pre = c[t][g] + b1[hcol];
      float gl  = 0.5f * pre * (1.0f + erff(pre * 0.70710678118654752f)); // exact gelu
      acc += gl * W2[hcol];
    }
    acc += __shfl_xor(acc, 1, 32);
    acc += __shfl_xor(acc, 2, 32);
    acc += __shfl_xor(acc, 4, 32);
    acc += __shfl_xor(acc, 8, 32);   // full sum within each 16-lane half
    if ((lane & 15) == 0) {
      int row = rowBase + g + half * 8;
      atomicAdd(&scores[row], acc);
    }
  }
}

// ---------------------------------------------------------------------------
// Kernel 4: per-batch top-k (k = S/2) via in-LDS bitonic sort of 4096
// (score, idx) pairs, descending by score with lower-index-first ties
// (matches jax.lax.top_k determinism). Also streams raw scores to d_out.
// ---------------------------------------------------------------------------
__global__ __launch_bounds__(1024) void k_topk(
    const float* __restrict__ scores,     // [B*S] (ws)
    float*       __restrict__ out_scores, // [B*S] (d_out section)
    float*       __restrict__ out_idx,    // [B*K] (d_out section, as float)
    int*         __restrict__ idx_ws)     // [B*K] (ws, for gather)
{
  __shared__ float s[SEQ];
  __shared__ int  ix[SEQ];
  const int b = blockIdx.x;
  const int t = threadIdx.x;

  for (int j = 0; j < 4; ++j) {
    int i    = t + j * 1024;
    float v  = scores[b * SEQ + i];
    s[i]     = v;
    ix[i]    = i;
    out_scores[b * SEQ + i] = v;
  }
  __syncthreads();

  for (int len = 2; len <= SEQ; len <<= 1) {
    for (int str = len >> 1; str > 0; str >>= 1) {
      for (int j = 0; j < 4; ++j) {
        int i = t + j * 1024;
        int p = i ^ str;
        if (p > i) {
          float si = s[i], sp = s[p];
          int   ii = ix[i], ip = ix[p];
          // "p's element should precede i's" in descending order:
          bool beforePI = (sp > si) || (sp == si && ip < ii);
          bool descBlk  = ((i & len) == 0);
          if (descBlk ? beforePI : !beforePI) {
            s[i] = sp; s[p] = si; ix[i] = ip; ix[p] = ii;
          }
        }
      }
      __syncthreads();
    }
  }

  for (int j = 0; j < 2; ++j) {
    int r  = t + j * 1024;     // r < TOPK
    int id = ix[r];
    out_idx[b * TOPK + r] = (float)id;
    idx_ws [b * TOPK + r] = id;
  }
}

// ---------------------------------------------------------------------------
// Kernel 5: gather selected rows: out[b,i,:] = x[b, idx[b,i], :], float4 copies.
// ---------------------------------------------------------------------------
__global__ __launch_bounds__(256) void k_gather(
    const float* __restrict__ x,
    const int*   __restrict__ idx,
    float*       __restrict__ out)
{
  int row = blockIdx.x;            // 0 .. B*TOPK-1
  int b   = row >> 11;             // TOPK = 2048
  int id  = idx[row];
  const float4* src = (const float4*)(x + ((size_t)b * SEQ + id) * D_MODEL);
  float4*       dst = (float4*)(out + (size_t)row * D_MODEL);
  for (int i = threadIdx.x; i < D_MODEL / 4; i += 256) dst[i] = src[i];
}

// ---------------------------------------------------------------------------
extern "C" void kernel_launch(void* const* d_in, const int* in_sizes, int n_in,
                              void* d_out, int out_size, void* d_ws, size_t ws_size,
                              hipStream_t stream) {
  (void)in_sizes; (void)n_in; (void)out_size; (void)ws_size;
  const float* x  = (const float*)d_in[0];
  const float* W1 = (const float*)d_in[1];
  const float* b1 = (const float*)d_in[2];
  const float* W2 = (const float*)d_in[3];
  const float* b2 = (const float*)d_in[4];

  float* out        = (float*)d_out;
  float* out_sel    = out;                                        // [B,K,D]
  float* out_idx    = out + (size_t)BATCH * TOPK * D_MODEL;       // [B,K]
  float* out_scores = out_idx + (size_t)BATCH * TOPK;             // [B,S]

  char*     ws     = (char*)d_ws;
  _Float16* W1T    = (_Float16*)ws;                                        // 2 MB
  float*    scores = (float*)(ws + (size_t)D_MODEL * HIDDEN * sizeof(_Float16));
  int*      idx_ws = (int*)((char*)scores + (size_t)BATCH * SEQ * sizeof(float));

  k_convert_w1   <<<(D_MODEL * HIDDEN + 255) / 256, 256, 0, stream>>>(W1, W1T);
  k_init_scores  <<<(BATCH * SEQ + 255) / 256,      256, 0, stream>>>(scores, b2);
  k_router_scores<<<BATCH * SEQ / 16,               256, 0, stream>>>(x, W1T, b1, W2, scores);
  k_topk         <<<BATCH,                         1024, 0, stream>>>(scores, out_scores, out_idx, idx_ws);
  k_gather       <<<BATCH * TOPK,                   256, 0, stream>>>(x, idx_ws, out_sel);
}
